// SpatialConv_13726715478759
// MI455X (gfx1250) — 8-bit matrix operand order repaired and checked by reading, compiled, not run on hardware
//
#include <hip/hip_runtime.h>


typedef __attribute__((ext_vector_type(4)))  int   v4i;
typedef __attribute__((ext_vector_type(8)))  int   v8i;
typedef __attribute__((ext_vector_type(16))) int   v16i;
typedef __attribute__((ext_vector_type(8)))  float v8f;

#define D_CH        1024
#define L_LEN       4096
#define K_TAPS      2048
#define N_BATCH     8
#define NCHUNK      17
#define ATILE_B     2048
#define A_TOTAL_B   (NCHUNK * ATILE_B)
#define XPAD_B      6272
#define XOFF        1024
#define KSCALE      4096.0f
#define KSCALE_INV  (1.0f / 4096.0f)

__device__ __forceinline__ unsigned sw_fp8_1(float f) {
    unsigned u = __float_as_uint(f);
    unsigned sgn = (u >> 24) & 0x80u;
    unsigned a = u & 0x7fffffffu;
    if (a >= 0x43e00000u) return sgn | 0x7eu;
    if (a <  0x3c800000u) return sgn;
    unsigned mant = a & 0x7fffffu, exp = a >> 23;
    unsigned keep = mant >> 20, rest = mant & 0xfffffu;
    keep += (rest > 0x80000u) || ((rest == 0x80000u) && (keep & 1u));
    unsigned e8 = exp - 127u + 7u;
    if (keep == 8u) { keep = 0u; e8 += 1u; }
    if (e8 >= 16u) return sgn | 0x7eu;
    return sgn | (e8 << 3) | keep;
}
__device__ __forceinline__ unsigned pack_fp8x4(float a, float b, float c, float d) {
#if __has_builtin(__builtin_amdgcn_cvt_pk_fp8_f32)
    int w = 0;
    w = __builtin_amdgcn_cvt_pk_fp8_f32(a, b, w, false);
    w = __builtin_amdgcn_cvt_pk_fp8_f32(c, d, w, true);
    return (unsigned)w;
#else
    return sw_fp8_1(a) | (sw_fp8_1(b) << 8) | (sw_fp8_1(c) << 16) | (sw_fp8_1(d) << 24);
#endif
}

__global__ __launch_bounds__(256)
void build_atiles_kernel(const float* __restrict__ kern, unsigned char* __restrict__ Aws) {
    int gid = blockIdx.x * blockDim.x + threadIdx.x;
    if (gid >= D_CH * NCHUNK * 32) return;
    int lane = gid & 31;
    int c    = (gid >> 5) % NCHUNK;
    int d    = gid / (NCHUNK * 32);
    int m = lane & 15, h = lane >> 4;
    const float* kd = kern + (size_t)d * K_TAPS;

    unsigned w[16];
    #pragma unroll
    for (int blk = 0; blk < 2; ++blk) {
        #pragma unroll
        for (int v = 0; v < 8; ++v) {
            int ub = 64 * blk + 32 * (v >> 2) + 16 * h + 4 * (v & 3);
            float f[4];
            #pragma unroll
            for (int j = 0; j < 4; ++j) {
                int t = 128 * c + m - (ub + j);
                f[j] = (t >= 0 && t < K_TAPS) ? kd[t] * KSCALE : 0.0f;
            }
            w[blk * 8 + v] = pack_fp8x4(f[0], f[1], f[2], f[3]);
        }
    }
    uint4* o = reinterpret_cast<uint4*>(Aws + ((size_t)(d * NCHUNK + c) * 32 + lane) * 64);
    o[0] = make_uint4(w[0],  w[1],  w[2],  w[3]);
    o[1] = make_uint4(w[4],  w[5],  w[6],  w[7]);
    o[2] = make_uint4(w[8],  w[9],  w[10], w[11]);
    o[3] = make_uint4(w[12], w[13], w[14], w[15]);
}

__device__ __forceinline__ v16i cat16(v4i q0, v4i q1, v4i q2, v4i q3) {
    v8i a = __builtin_shufflevector(q0, q1, 0,1,2,3,4,5,6,7);
    v8i b = __builtin_shufflevector(q2, q3, 0,1,2,3,4,5,6,7);
    return __builtin_shufflevector(a, b, 0,1,2,3,4,5,6,7,8,9,10,11,12,13,14,15);
}
__device__ __forceinline__ v16i loadA64(const unsigned char* p) {
    const v4i* q = reinterpret_cast<const v4i*>(p);
    return cat16(q[0], q[1], q[2], q[3]);
}
__device__ __forceinline__ v16i loadB64(const unsigned char* xp, int colbase, int h) {
    v4i q0 = *reinterpret_cast<const v4i*>(xp + colbase + 16 * h);
    v4i q1 = *reinterpret_cast<const v4i*>(xp + colbase + 16 * h + 32);
    v4i q2 = *reinterpret_cast<const v4i*>(xp + colbase + 16 * h + 64);
    v4i q3 = *reinterpret_cast<const v4i*>(xp + colbase + 16 * h + 96);
    return cat16(q0, q1, q2, q3);
}

__global__ __launch_bounds__(256)
void conv_wmma_kernel(const float* __restrict__ x, const unsigned char* __restrict__ Aws,
                      const float* __restrict__ bias, float* __restrict__ out) {
    __shared__ __align__(16) unsigned char xpad[XPAD_B];
    __shared__ __align__(16) unsigned char atile[A_TOTAL_B];

    const int b   = blockIdx.x >> 10;
    const int d   = blockIdx.x & (D_CH - 1);
    const int tid = threadIdx.x;

    const unsigned char* Asrc = Aws + (size_t)d * A_TOTAL_B;
    for (int j = tid; j < A_TOTAL_B / 16; j += 256) {
        unsigned long long ga = (unsigned long long)(uintptr_t)(Asrc + (size_t)j * 16);
        unsigned int       la = (unsigned int)(uintptr_t)(atile + (size_t)j * 16);
        asm volatile("global_load_async_to_lds_b128 %0, %1, off"
                     :: "v"(la), "v"(ga) : "memory");
    }

    {
        uint4 z = make_uint4(0, 0, 0, 0);
        uint4* xv = reinterpret_cast<uint4*>(xpad);
        for (int j = tid; j < XPAD_B / 16; j += 256)
            if (j < XOFF / 16 || j >= (XOFF + L_LEN) / 16) xv[j] = z;
    }

    const float* xg = x + ((size_t)b * D_CH + d) * L_LEN;
    #pragma unroll
    for (int k = 0; k < 2; ++k) {
        int i8 = tid + 256 * k;
        float4 va = reinterpret_cast<const float4*>(xg)[i8 * 2];
        float4 vb = reinterpret_cast<const float4*>(xg)[i8 * 2 + 1];
        uint2 pk;
        pk.x = pack_fp8x4(va.x, va.y, va.z, va.w);
        pk.y = pack_fp8x4(vb.x, vb.y, vb.z, vb.w);
        *reinterpret_cast<uint2*>(&xpad[XOFF + i8 * 8]) = pk;
    }

    asm volatile("s_wait_asynccnt 0x0" ::: "memory");
    __syncthreads();

    const int lane = tid & 31;
    const int wave = tid >> 5;
    const int nn   = lane & 15;
    const int h    = lane >> 4;
    const int p0a  = wave * 512;
    const int bbase = XOFF + p0a + 16 * nn + 1024;

    v8f acc0 = {}; v8f acc1 = {};
    const unsigned char* Ab = atile + lane * 64;

    v16i fifo[2];
    fifo[0] = loadB64(xpad, bbase + 256, h);
    fifo[1] = loadB64(xpad, bbase + 128, h);

    #pragma unroll
    for (int c = 0; c < NCHUNK; ++c) {
        v16i af = loadA64(Ab + c * ATILE_B);
        v16i bc = loadB64(xpad, bbase - 128 * c, h);
        acc0 = __builtin_amdgcn_wmma_f32_16x16x128_fp8_fp8(
                   af, bc, (short)0, acc0, false, false);
        v16i bo = fifo[c & 1];
        acc1 = __builtin_amdgcn_wmma_f32_16x16x128_fp8_fp8(
                   af, bo, (short)0, acc1, false, false);
        fifo[c & 1] = bc;
    }

    const float bv = bias[d];
    float* outp = out + ((size_t)b * D_CH + d) * L_LEN;
    #pragma unroll
    for (int r = 0; r < 8; ++r) {
        outp[p0a + r + 8 * h + 16 * nn]       = acc0[r] * KSCALE_INV + bv;
        outp[p0a + 256 + r + 8 * h + 16 * nn] = acc1[r] * KSCALE_INV + bv;
    }
}

extern "C" void kernel_launch(void* const* d_in, const int* in_sizes, int n_in,
                              void* d_out, int out_size, void* d_ws, size_t ws_size,
                              hipStream_t stream) {
    const float* x    = (const float*)d_in[0];
    const float* kern = (const float*)d_in[1];
    const float* bias = (const float*)d_in[2];
    float* out = (float*)d_out;
    unsigned char* Aws = (unsigned char*)d_ws;

    (void)in_sizes; (void)n_in; (void)out_size; (void)ws_size;

    int prep_threads = D_CH * NCHUNK * 32;
    build_atiles_kernel<<<(prep_threads + 255) / 256, 256, 0, stream>>>(kern, Aws);
    conv_wmma_kernel<<<N_BATCH * D_CH, 256, 0, stream>>>(x, Aws, bias, out);
}
